// BiMamba3Block_43636867728066
// MI455X (gfx1250) — matrix operand order repaired and checked by reading, not compiled, not run on hardware
//
#include <hip/hip_runtime.h>
#include <hip/hip_bf16.h>
#include <math.h>

// ---------------- problem dims (from reference setup_inputs) ----------------
#define BSZ     2
#define LEN     1024
#define DMODEL  1024
#define CONDSZ  1024
#define DINNER  2048
#define NSTATE  16
#define RLOW    64
#define KCONV   4
#define HIDSZ   2048
#define MROWS   (BSZ*LEN)         // 2048 token rows
#define DBCW    (RLOW + 2*NSTATE) // 96
#define DBCP    128               // padded x-proj weight width (multiple of 64)

typedef __attribute__((ext_vector_type(16))) __bf16 v16bf;
typedef __attribute__((ext_vector_type(8)))  __bf16 v8bf;
typedef __attribute__((ext_vector_type(8)))  float  v8f;

__device__ __forceinline__ __bf16 tobf(float f) {
    __hip_bfloat16 h = __float2bfloat16(f);
    return *reinterpret_cast<__bf16*>(&h);
}
__device__ __forceinline__ float silu(float v) {
    return v / (1.0f + __expf(-v));
}

// ---------------- elementwise / conversion kernels ----------------
__global__ void k_cvt_bf16(const float* __restrict__ s, __bf16* __restrict__ d, int n) {
    int i = blockIdx.x * blockDim.x + threadIdx.x;
    if (i < n) d[i] = tobf(s[i]);
}

// convert KxN f32 weight into KxNp bf16 (zero-padded columns N..Np-1)
__global__ void k_cvt_bf16_pad(const float* __restrict__ s, __bf16* __restrict__ d,
                               int N, int Np, int total) {
    int i = blockIdx.x * blockDim.x + threadIdx.x;
    if (i >= total) return;                  // total = K*Np
    int k = i / Np, n = i - k * Np;
    d[i] = (n < N) ? tobf(s[k * N + n]) : tobf(0.0f);
}

// convert (B, L, RS) rows to bf16, optionally time-flipped (for backward SSM)
__global__ void k_cvt_bf16_rows(const float* __restrict__ s, __bf16* __restrict__ d,
                                int RS, int rev) {
    int i = blockIdx.x * blockDim.x + threadIdx.x;
    int n = MROWS * RS;
    if (i >= n) return;
    int row = i / RS, j = i - row * RS;
    int b = row / LEN, l = row - b * LEN;
    int lo = rev ? (LEN - 1 - l) : l;
    d[(b * LEN + lo) * RS + j] = tobf(s[i]);
}

// mod = c @ W + bias   (B x 3D)
__global__ void k_condmod(const float* __restrict__ c, const float* __restrict__ W,
                          const float* __restrict__ bias, float* __restrict__ mod) {
    int j = blockIdx.x * blockDim.x + threadIdx.x;
    int b = blockIdx.y;
    if (j >= 3 * DMODEL) return;
    float acc = bias[j];
    const float* cr = c + b * CONDSZ;
    for (int k = 0; k < CONDSZ; ++k) acc += cr[k] * W[k * 3 * DMODEL + j];
    mod[b * 3 * DMODEL + j] = acc;
}

// adaLN: h = normed(x_row) * (1 + scale) + shift  (one 256-thread block per row)
__global__ void k_adaln(const float* __restrict__ x, const float* __restrict__ mod,
                        float* __restrict__ h) {
    int row = blockIdx.x;
    int b = row / LEN;
    __shared__ float red0[8], red1[8];
    float s = 0.f, ss = 0.f;
    for (int d = threadIdx.x; d < DMODEL; d += blockDim.x) {
        float v = x[row * DMODEL + d];
        s += v; ss += v * v;
    }
    for (int o = 16; o >= 1; o >>= 1) { s += __shfl_xor(s, o, 32); ss += __shfl_xor(ss, o, 32); }
    int wid = threadIdx.x >> 5, lane = threadIdx.x & 31;
    if (lane == 0) { red0[wid] = s; red1[wid] = ss; }
    __syncthreads();
    if (wid == 0) {
        s  = (lane < 8) ? red0[lane] : 0.f;
        ss = (lane < 8) ? red1[lane] : 0.f;
        for (int o = 4; o >= 1; o >>= 1) { s += __shfl_xor(s, o, 32); ss += __shfl_xor(ss, o, 32); }
        if (lane == 0) { red0[0] = s; red1[0] = ss; }
    }
    __syncthreads();
    float mu  = red0[0] * (1.0f / DMODEL);
    float var = red1[0] * (1.0f / DMODEL) - mu * mu;
    float rinv = rsqrtf(var + 1e-5f);
    const float* m = mod + b * 3 * DMODEL;
    for (int d = threadIdx.x; d < DMODEL; d += blockDim.x) {
        float v = (x[row * DMODEL + d] - mu) * rinv;
        h[row * DMODEL + d] = v * (1.f + m[DMODEL + d]) + m[d];
    }
}

// ---------------- WMMA bf16 GEMM: C[MxN] = A[MxK] @ W[K x ldw] ----------------
// W is pre-padded to ldw columns (multiple of 64) so tile loads are unguarded.
// 128 threads = 4 waves. Block tile: 32 (M) x 64 (N). Each wave owns two 16x16
// C tiles stacked in M, sharing one B fragment (2 v_wmma per K-step).
// W K-tile (32 x 64) staged TRANSPOSED in double-buffered LDS (wt[buf][n][k],
// padded K-stride) -> B frag = 2x ds_load_b128; A frag = 2x global b128.
// Single barrier per K-step; next tile's global loads overlap current WMMAs.
#define GTN 64
#define GTM 32
#define KP  40   // padded K stride in LDS elems (multiple of 8 keeps 16B align)

__global__ void k_gemm_bf16(const __bf16* __restrict__ A, const __bf16* __restrict__ W,
                            float* __restrict__ C, int M, int N, int K, int ldw) {
    __shared__ __bf16 wt[2][GTN * KP];       // 2 x 5120 bytes
    int lane = threadIdx.x & 31, wave = threadIdx.x >> 5;
    int m0   = blockIdx.x * GTM;
    int nblk = blockIdx.y * GTN;
    int n0   = nblk + wave * 16;
    int lh   = lane >> 4;                    // lane group
    int lm   = lane & 15;
    v8f acc0 = {0.f,0.f,0.f,0.f,0.f,0.f,0.f,0.f};
    v8f acc1 = {0.f,0.f,0.f,0.f,0.f,0.f,0.f,0.f};
    const bool active = (n0 < N);            // wave-uniform (N multiple of 16)

    // per-thread loader slots: slot0 = tid, slot1 = tid+128 -> kk in {t>>3, 16+t>>3}
    int kk0  = threadIdx.x >> 3;             // 0..15
    int nn0  = (threadIdx.x & 7) << 3;       // 0,8,..,56
    const __bf16* wp0 = W + (size_t)kk0 * ldw + nblk + nn0;
    const __bf16* wp1 = wp0 + (size_t)16 * ldw;
    int ds0 = nn0 * KP + kk0;                // transposed LDS slot bases
    int ds1 = nn0 * KP + kk0 + 16;
    const __bf16* Ar0 = A + (size_t)(m0 + lm) * K + lh * 8;
    const __bf16* Ar1 = A + (size_t)(m0 + 16 + lm) * K + lh * 8;
    const int KT = K >> 5;

    // prologue: stage tile 0 into buffer 0
    {
        v8bf r0 = *(const v8bf*)wp0;
        v8bf r1 = *(const v8bf*)wp1;
        #pragma unroll
        for (int j = 0; j < 8; ++j) {
            wt[0][ds0 + j * KP] = r0[j];
            wt[0][ds1 + j * KP] = r1[j];
        }
    }
    __syncthreads();

    for (int kt = 0; kt < KT; ++kt) {
        const int cur = kt & 1;
        const bool more = (kt + 1 < KT);
        v8bf r0, r1;
        if (more) {                          // issue next tile's global loads early
            const __bf16* q = wp0 + (size_t)(kt + 1) * 32 * ldw;
            r0 = *(const v8bf*)q;
            r1 = *(const v8bf*)(q + (size_t)16 * ldw);
        }
        if (active) {
            const int ko = kt * 32;
            v8bf a0lo = *(const v8bf*)(Ar0 + ko);
            v8bf a0hi = *(const v8bf*)(Ar0 + ko + 16);
            v8bf a1lo = *(const v8bf*)(Ar1 + ko);
            v8bf a1hi = *(const v8bf*)(Ar1 + ko + 16);
            const __bf16* Bc = &wt[cur][(wave * 16 + lm) * KP + lh * 8];
            v8bf blo = *(const v8bf*)(Bc);
            v8bf bhi = *(const v8bf*)(Bc + 16);
            v16bf af0 = __builtin_shufflevector(a0lo, a0hi,
                0,1,2,3,4,5,6,7,8,9,10,11,12,13,14,15);
            v16bf af1 = __builtin_shufflevector(a1lo, a1hi,
                0,1,2,3,4,5,6,7,8,9,10,11,12,13,14,15);
            v16bf bfv = __builtin_shufflevector(blo, bhi,
                0,1,2,3,4,5,6,7,8,9,10,11,12,13,14,15);
            if (more) __builtin_prefetch(Ar0 + ko + 32, 0, 1);  // global_prefetch
            acc0 = __builtin_amdgcn_wmma_f32_16x16x32_bf16(
                false, af0, false, bfv, (short)0, acc0, false, false);
            acc1 = __builtin_amdgcn_wmma_f32_16x16x32_bf16(
                false, af1, false, bfv, (short)0, acc1, false, false);
        }
        if (more) {                          // scatter staged regs into other buffer
            #pragma unroll
            for (int j = 0; j < 8; ++j) {
                wt[1 - cur][ds0 + j * KP] = r0[j];
                wt[1 - cur][ds1 + j * KP] = r1[j];
            }
        }
        __syncthreads();
    }
    if (active) {
        int col = n0 + lm;
        #pragma unroll
        for (int v = 0; v < 8; ++v) {        // f32 C/D layout: vgpr v -> M = v + lh*8
            C[(size_t)(m0 + v + lh * 8) * N + col]      = acc0[v];
            C[(size_t)(m0 + 16 + v + lh * 8) * N + col] = acc1[v];
        }
    }
}

// causal depthwise conv (K=4) + bias + silu; writes u (f32) and u (bf16)
__global__ void k_conv_silu(const float* __restrict__ xz, const float* __restrict__ cw,
                            const float* __restrict__ cb, float* __restrict__ u,
                            __bf16* __restrict__ ubf) {
    int i = blockIdx.x * blockDim.x + threadIdx.x;
    int n = MROWS * DINNER;
    if (i >= n) return;
    int d = i % DINNER;
    int row = i / DINNER;
    int b = row / LEN, l = row - b * LEN;
    float acc = cb[d];
    #pragma unroll
    for (int j = 0; j < KCONV; ++j) {
        int ls = l - (KCONV - 1) + j;
        if (ls >= 0) acc += cw[d * KCONV + j] * xz[(b * LEN + ls) * (2 * DINNER) + d];
    }
    float s = silu(acc);
    u[i] = s;
    ubf[i] = tobf(s);
}

// slice dt_r (cols 0..R-1 of dbc) to bf16
__global__ void k_dtr_bf16(const float* __restrict__ dbc, __bf16* __restrict__ dtr) {
    int i = blockIdx.x * blockDim.x + threadIdx.x;
    if (i >= MROWS * RLOW) return;
    int row = i / RLOW, j = i - row * RLOW;
    dtr[i] = tobf(dbc[row * DBCW + j]);
}

// dt = softplus(dt_raw + dt_b)  (in place)
__global__ void k_dtact(float* __restrict__ dt, const float* __restrict__ dtb) {
    int i = blockIdx.x * blockDim.x + threadIdx.x;
    if (i >= MROWS * DINNER) return;
    int d = i % DINNER;
    float v = dt[i] + dtb[d];
    dt[i] = (v > 20.f) ? v : log1pf(__expf(v));
}

// selective scan: one (b,d) channel per 16-lane half-wave; N=16 states on lanes.
__global__ void k_scan(const float* __restrict__ dt, const float* __restrict__ u,
                       const float* __restrict__ dbc, const float* __restrict__ Alog,
                       const float* __restrict__ Dv, float* __restrict__ y) {
    int half = threadIdx.x >> 4;                // 0..15 within the 256-thread block
    int ch = blockIdx.x * 16 + half;            // 0 .. B*dI-1
    int n = threadIdx.x & 15;
    int b = ch / DINNER, d = ch - b * DINNER;
    float A = -__expf(Alog[d * NSTATE + n]);
    float Dd = Dv[d];
    float h = 0.f;
    for (int l = 0; l < LEN; ++l) {
        int row = b * LEN + l;
        float dtv = dt[row * DINNER + d];
        float uv  = u[row * DINNER + d];
        float Bv  = dbc[row * DBCW + RLOW + n];
        float Cv  = dbc[row * DBCW + RLOW + NSTATE + n];
        h = h * __expf(dtv * A) + dtv * uv * Bv;
        float p = h * Cv;
        p += __shfl_xor(p, 8, 32);
        p += __shfl_xor(p, 4, 32);
        p += __shfl_xor(p, 2, 32);
        p += __shfl_xor(p, 1, 32);
        if (n == 0) y[row * DINNER + d] = p + Dd * uv;
    }
}

// y2 = y * silu(z), bf16, with optional time un-flip (for backward dir)
__global__ void k_ygate(const float* __restrict__ y, const float* __restrict__ xz,
                        __bf16* __restrict__ ybf, int rev) {
    int i = blockIdx.x * blockDim.x + threadIdx.x;
    if (i >= MROWS * DINNER) return;
    int d = i % DINNER;
    int row = i / DINNER;
    int b = row / LEN, l = row - b * LEN;
    float z = xz[row * (2 * DINNER) + DINNER + d];
    float v = y[i] * silu(z);
    int lo = rev ? (LEN - 1 - l) : l;
    ybf[(b * LEN + lo) * DINNER + d] = tobf(v);
}

__global__ void k_res2(const float* __restrict__ x, const float* __restrict__ a,
                       const float* __restrict__ bb, const float* __restrict__ mod,
                       float* __restrict__ out) {
    int i = blockIdx.x * blockDim.x + threadIdx.x;
    if (i >= MROWS * DMODEL) return;
    int d = i % DMODEL;
    int b = (i / DMODEL) / LEN;
    float g = mod[b * 3 * DMODEL + 2 * DMODEL + d];
    out[i] = x[i] + g * (a[i] + bb[i]);
}

__global__ void k_res1(const float* __restrict__ x, const float* __restrict__ a,
                       const float* __restrict__ mod, float* __restrict__ out) {
    int i = blockIdx.x * blockDim.x + threadIdx.x;
    if (i >= MROWS * DMODEL) return;
    int d = i % DMODEL;
    int b = (i / DMODEL) / LEN;
    float g = mod[b * 3 * DMODEL + 2 * DMODEL + d];
    out[i] = x[i] + g * a[i];
}

__global__ void k_swiglu(const float* __restrict__ a1, const float* __restrict__ a2,
                         __bf16* __restrict__ g) {
    int i = blockIdx.x * blockDim.x + threadIdx.x;
    if (i >= MROWS * HIDSZ) return;
    g[i] = tobf(silu(a1[i]) * a2[i]);
}

// ---------------- host orchestration ----------------
static inline int ceil_div(int a, int b) { return (a + b - 1) / b; }

extern "C" void kernel_launch(void* const* d_in, const int* in_sizes, int n_in,
                              void* d_out, int out_size, void* d_ws, size_t ws_size,
                              hipStream_t stream) {
    (void)in_sizes; (void)n_in; (void)out_size; (void)ws_size;
    const float* x = (const float*)d_in[0];
    const float* c = (const float*)d_in[1];
    // ssm param dict order: in_W, conv_w, conv_b, xproj_W, dt_W, dt_b, A_log, D, out_W
    const float* fw[9]; const float* bw[9];
    for (int i = 0; i < 9; ++i) { fw[i] = (const float*)d_in[2 + i]; bw[i] = (const float*)d_in[11 + i]; }
    const float* adaln1_W = (const float*)d_in[20];
    const float* adaln1_b = (const float*)d_in[21];
    const float* adaln2_W = (const float*)d_in[22];
    const float* adaln2_b = (const float*)d_in[23];
    const float* mlp_w1   = (const float*)d_in[24];
    const float* mlp_w2   = (const float*)d_in[25];
    const float* mlp_w3   = (const float*)d_in[26];

    // ---- workspace carve-up (256B aligned) ----
    char* base = (char*)d_ws;
    size_t off = 0;
    auto carve = [&](size_t bytes) -> char* {
        char* p = base + off;
        off = (off + bytes + 255) & ~(size_t)255;
        return p;
    };
    __bf16* WBF  = (__bf16*)carve((size_t)DMODEL * 2 * DINNER * 2); // largest weight, reused for all
    __bf16* HBF  = (__bf16*)carve((size_t)MROWS * DMODEL * 2);
    __bf16* UBF  = (__bf16*)carve((size_t)MROWS * DINNER * 2);
    __bf16* DTRB = (__bf16*)carve((size_t)MROWS * RLOW * 2);
    __bf16* YBF  = (__bf16*)carve((size_t)MROWS * DINNER * 2);      // reused as SwiGLU bf16 buf
    float*  MOD1 = (float*)carve((size_t)BSZ * 3 * DMODEL * 4);
    float*  MOD2 = (float*)carve((size_t)BSZ * 3 * DMODEL * 4);
    float*  H    = (float*)carve((size_t)MROWS * DMODEL * 4);
    float*  XZ   = (float*)carve((size_t)MROWS * 2 * DINNER * 4);   // MLP a1/a2 alias here
    float*  U    = (float*)carve((size_t)MROWS * DINNER * 4);
    float*  DBC  = (float*)carve((size_t)MROWS * DBCW * 4);
    float*  DT   = (float*)carve((size_t)MROWS * DINNER * 4);
    float*  Y    = (float*)carve((size_t)MROWS * DINNER * 4);       // MLP out aliases here
    float*  OF   = (float*)carve((size_t)MROWS * DMODEL * 4);
    float*  OB   = (float*)carve((size_t)MROWS * DMODEL * 4);
    float*  X1   = (float*)carve((size_t)MROWS * DMODEL * 4);
    float*  A1   = XZ;
    float*  A2   = XZ + (size_t)MROWS * HIDSZ;
    float*  MLPO = Y;

    const int TB = 256;
    dim3 cm_grid(ceil_div(3 * DMODEL, TB), BSZ);
    k_condmod<<<cm_grid, TB, 0, stream>>>(c, adaln1_W, adaln1_b, MOD1);
    k_condmod<<<cm_grid, TB, 0, stream>>>(c, adaln2_W, adaln2_b, MOD2);
    k_adaln<<<MROWS, TB, 0, stream>>>(x, MOD1, H);

    for (int dir = 0; dir < 2; ++dir) {
        const float* const* p = dir ? bw : fw;
        float* OD = dir ? OB : OF;
        k_cvt_bf16_rows<<<ceil_div(MROWS * DMODEL, TB), TB, 0, stream>>>(H, HBF, DMODEL, dir);
        // xz = h @ in_W
        k_cvt_bf16<<<ceil_div(DMODEL * 2 * DINNER, TB), TB, 0, stream>>>(p[0], WBF, DMODEL * 2 * DINNER);
        k_gemm_bf16<<<dim3(MROWS / GTM, (2 * DINNER) / GTN), 128, 0, stream>>>(
            HBF, WBF, XZ, MROWS, 2 * DINNER, DMODEL, 2 * DINNER);
        // u = silu(causal_conv(x_in) + b)
        k_conv_silu<<<ceil_div(MROWS * DINNER, TB), TB, 0, stream>>>(XZ, p[1], p[2], U, UBF);
        // dbc = u @ xproj_W  (weight zero-padded 96 -> 128 cols)
        k_cvt_bf16_pad<<<ceil_div(DINNER * DBCP, TB), TB, 0, stream>>>(
            p[3], WBF, DBCW, DBCP, DINNER * DBCP);
        k_gemm_bf16<<<dim3(MROWS / GTM, DBCP / GTN), 128, 0, stream>>>(
            UBF, WBF, DBC, MROWS, DBCW, DINNER, DBCP);
        // dt = softplus(dt_r @ dt_W + dt_b)
        k_dtr_bf16<<<ceil_div(MROWS * RLOW, TB), TB, 0, stream>>>(DBC, DTRB);
        k_cvt_bf16<<<ceil_div(RLOW * DINNER, TB), TB, 0, stream>>>(p[4], WBF, RLOW * DINNER);
        k_gemm_bf16<<<dim3(MROWS / GTM, DINNER / GTN), 128, 0, stream>>>(
            DTRB, WBF, DT, MROWS, DINNER, RLOW, DINNER);
        k_dtact<<<ceil_div(MROWS * DINNER, TB), TB, 0, stream>>>(DT, p[5]);
        // selective scan (fp32, sequential over L)
        k_scan<<<(BSZ * DINNER) / 16, TB, 0, stream>>>(DT, U, DBC, p[6], p[7], Y);
        // gate by silu(z), un-flip for backward dir, to bf16
        k_ygate<<<ceil_div(MROWS * DINNER, TB), TB, 0, stream>>>(Y, XZ, YBF, dir);
        // dir_out = y2 @ out_W
        k_cvt_bf16<<<ceil_div(DINNER * DMODEL, TB), TB, 0, stream>>>(p[8], WBF, DINNER * DMODEL);
        k_gemm_bf16<<<dim3(MROWS / GTM, DMODEL / GTN), 128, 0, stream>>>(
            YBF, WBF, OD, MROWS, DMODEL, DINNER, DMODEL);
    }

    // x1 = x + gate1 * (fwd + bwd)
    k_res2<<<ceil_div(MROWS * DMODEL, TB), TB, 0, stream>>>(x, OF, OB, MOD1, X1);

    // MLP branch
    k_adaln<<<MROWS, TB, 0, stream>>>(X1, MOD2, H);
    k_cvt_bf16_rows<<<ceil_div(MROWS * DMODEL, TB), TB, 0, stream>>>(H, HBF, DMODEL, 0);
    k_cvt_bf16<<<ceil_div(DMODEL * HIDSZ, TB), TB, 0, stream>>>(mlp_w1, WBF, DMODEL * HIDSZ);
    k_gemm_bf16<<<dim3(MROWS / GTM, HIDSZ / GTN), 128, 0, stream>>>(
        HBF, WBF, A1, MROWS, HIDSZ, DMODEL, HIDSZ);
    k_cvt_bf16<<<ceil_div(DMODEL * HIDSZ, TB), TB, 0, stream>>>(mlp_w2, WBF, DMODEL * HIDSZ);
    k_gemm_bf16<<<dim3(MROWS / GTM, HIDSZ / GTN), 128, 0, stream>>>(
        HBF, WBF, A2, MROWS, HIDSZ, DMODEL, HIDSZ);
    k_swiglu<<<ceil_div(MROWS * HIDSZ, TB), TB, 0, stream>>>(A1, A2, YBF);
    k_cvt_bf16<<<ceil_div(HIDSZ * DMODEL, TB), TB, 0, stream>>>(mlp_w3, WBF, HIDSZ * DMODEL);
    k_gemm_bf16<<<dim3(MROWS / GTM, DMODEL / GTN), 128, 0, stream>>>(
        YBF, WBF, MLPO, MROWS, DMODEL, HIDSZ, DMODEL);

    // out = x1 + gate2 * mlp_out
    k_res1<<<ceil_div(MROWS * DMODEL, TB), TB, 0, stream>>>(X1, MLPO, MOD2, (float*)d_out);
}
